// TransformerLayer_1752346656957
// MI455X (gfx1250) — matrix operand order repaired and checked by reading, not compiled, not run on hardware
//
#include <hip/hip_runtime.h>
#include <hip/hip_bf16.h>

// ---------------------------------------------------------------------------
// Types for CDNA5 WMMA (gfx1250, wave32)
// ---------------------------------------------------------------------------
typedef __attribute__((ext_vector_type(16))) __bf16 v16bf;
typedef __attribute__((ext_vector_type(8)))  float  v8f;

union Frag16 {
    v16bf        bf;
    uint4        q[2];     // two 16-byte halves (8 bf16 each)
    unsigned int u[8];
};

__device__ __forceinline__ unsigned short f2bf(float f) {
    union { float f; unsigned int u; } c; c.f = f;
    unsigned int u = c.u;
    unsigned int r = u + 0x7fffu + ((u >> 16) & 1u);   // round-to-nearest-even
    return (unsigned short)(r >> 16);
}
__device__ __forceinline__ unsigned int pack2(float lo, float hi) {
    return (unsigned int)f2bf(lo) | ((unsigned int)f2bf(hi) << 16);
}
__device__ __forceinline__ v8f wmma_bf16(const Frag16& a, const Frag16& b, v8f c) {
    return __builtin_amdgcn_wmma_f32_16x16x32_bf16(
        false, a.bf, false, b.bf, (short)0, c, false, false);
}

// ---- CDNA5 async global->LDS copy (ASYNCcnt-tracked, bypasses VGPRs) ----
__device__ __forceinline__ void async_copy_b128(unsigned lds_off, const void* gsrc) {
    asm volatile("global_load_async_to_lds_b128 %0, %1, off"
                 :: "v"(lds_off), "v"(gsrc)
                 : "memory");
}
__device__ __forceinline__ void wait_asynccnt0() {
#if __has_builtin(__builtin_amdgcn_s_wait_asynccnt)
    __builtin_amdgcn_s_wait_asynccnt(0);
#else
    asm volatile("s_wait_asynccnt 0x0" ::: "memory");
#endif
}
__device__ __forceinline__ unsigned lds_offset(const void* p) {
    return (unsigned)(unsigned long long)p;   // flat->LDS aperture uses addr[31:0]
}

#define BB    4
#define SS    1024
#define DD    1024
#define HH    16
#define DKK   64
#define DFFN  4096
#define LREL  128
#define NREL  257            // 2*L+1
#define NRELP 288            // padded to multiple of 32 for WMMA K loop
#define HD    1024           // H*DK

// ---------------------------------------------------------------------------
// Elementwise f32 -> bf16 convert
// ---------------------------------------------------------------------------
__global__ __launch_bounds__(256)
void cvt_bf16_kernel(const float* __restrict__ in, unsigned short* __restrict__ out, long n)
{
    const long i = (long)blockIdx.x * 256 + threadIdx.x;
    if (i < n) out[i] = f2bf(in[i]);
}

// W_rel [257,64] f32 -> transposed, padded bf16 [64, 288]
__global__ __launch_bounds__(256)
void wrelT_kernel(const float* __restrict__ wrel, unsigned short* __restrict__ wrelT)
{
    const int i = blockIdx.x * 256 + threadIdx.x;
    if (i < DKK * NRELP) {
        const int d = i / NRELP, rel = i % NRELP;
        wrelT[i] = (rel < NREL) ? f2bf(wrel[(long)rel * DKK + d]) : (unsigned short)0;
    }
}

// V [B,S,H*DK] f32 -> bf16 transposed vT [B,H,DK,S]  (k-major for P@V B-frags)
__global__ __launch_bounds__(256)
void vT_kernel(const float* __restrict__ v, unsigned short* __restrict__ vT)
{
    const long row = blockIdx.x;                 // (b*H + h)*DK + d
    const int  d   = (int)(row % DKK);
    const long bh  = row / DKK;
    const int  h   = (int)(bh % HH);
    const int  b   = (int)(bh / HH);
    for (int s = threadIdx.x; s < SS; s += 256)
        vT[row * SS + s] = f2bf(v[((long)(b * SS + s)) * HD + h * DKK + d]);
}

// ---------------------------------------------------------------------------
// Register-blocked bf16 GEMM: C[M,N] = A[M,K] @ W[N,K]^T + bias (+ReLU).
// Block 256 threads = 8 waves (2 M x 4 N). Block tile 64x128, wave tile 32x32
// (4 accumulators, 4 WMMAs per k-step). A/B staged via async global->LDS.
// ---------------------------------------------------------------------------
__global__ __launch_bounds__(256)
void gemm_bf16_kernel(const unsigned short* __restrict__ A,
                      const unsigned short* __restrict__ W,
                      const float* __restrict__ bias,
                      float* __restrict__ Cf, unsigned short* __restrict__ Cbf,
                      int M, int N, int K, int relu)
{
    __shared__ unsigned short a_sm[64][32];      // 4 KB
    __shared__ unsigned short b_sm[128][32];     // 8 KB

    const int tid  = threadIdx.x;
    const int lane = tid & 31;
    const int wave = tid >> 5;
    const int half = lane >> 4;
    const int l15  = lane & 15;
    const int wm   = wave >> 2;                  // 0..1  (M direction)
    const int wn   = wave & 3;                   // 0..3  (N direction)

    const long m0 = (long)blockIdx.y * 64;
    const long n0 = (long)blockIdx.x * 128;

    // staging maps (all b128)
    const int sa_row = tid >> 2;                 // 0..63
    const int sa_j   = (tid & 3) * 8;            // 0,8,16,24
    const int sb_row = tid >> 1;                 // 0..127
    const int sb_j   = (tid & 1) * 16;           // 0,16

    const unsigned a_dst  = lds_offset(&a_sm[sa_row][sa_j]);
    const unsigned b_dst0 = lds_offset(&b_sm[sb_row][sb_j]);
    const unsigned b_dst1 = lds_offset(&b_sm[sb_row][sb_j + 8]);
    const bool     b_ok   = (n0 + sb_row) < (long)N;

    v8f acc[2][2] = {};

    for (int k0 = 0; k0 < K; k0 += 32) {
        // ---- async-stage A 64x32 tile (ASYNCcnt, no VGPR bounce) ----
        async_copy_b128(a_dst, A + (m0 + sa_row) * (long)K + k0 + sa_j);
        // ---- async-stage B 128x32 tile (k-major rows of W) ----
        if (b_ok) {
            const unsigned short* wp = W + (n0 + sb_row) * (long)K + k0 + sb_j;
            __builtin_prefetch(wp + 32, 0, 3);
            async_copy_b128(b_dst0, wp);
            async_copy_b128(b_dst1, wp + 8);
        } else {
            const uint4 z = {0u, 0u, 0u, 0u};
            *(uint4*)&b_sm[sb_row][sb_j]     = z;
            *(uint4*)&b_sm[sb_row][sb_j + 8] = z;
        }
        wait_asynccnt0();
        __syncthreads();

        // ---- fragments from LDS ----
        Frag16 af[2], bfr[2];
#pragma unroll
        for (int t = 0; t < 2; ++t) {
            const int ar = wm * 32 + t * 16 + l15;
            af[t].q[0] = *(const uint4*)&a_sm[ar][16 * half];
            af[t].q[1] = *(const uint4*)&a_sm[ar][16 * half + 8];
            const int br = wn * 32 + t * 16 + l15;
            bfr[t].q[0] = *(const uint4*)&b_sm[br][16 * half];
            bfr[t].q[1] = *(const uint4*)&b_sm[br][16 * half + 8];
        }

        // ---- 4 WMMAs per k-step (2x2 register block) ----
#pragma unroll
        for (int i = 0; i < 2; ++i)
#pragma unroll
            for (int j = 0; j < 2; ++j)
                acc[i][j] = wmma_bf16(af[i], bfr[j], acc[i][j]);
        __syncthreads();
    }

    // ---- epilogue ----
#pragma unroll
    for (int j = 0; j < 2; ++j) {
        const long ncol = n0 + wn * 32 + j * 16 + l15;
        if (ncol >= (long)N) continue;
        const float bv = bias ? bias[ncol] : 0.f;
#pragma unroll
        for (int i = 0; i < 2; ++i) {
#pragma unroll
            for (int r = 0; r < 8; ++r) {
                float vv = acc[i][j][r] + bv;
                if (relu) vv = fmaxf(vv, 0.f);
                const long idx =
                    (m0 + wm * 32 + i * 16 + r + 8 * half) * (long)N + ncol;
                if (Cf)  Cf[idx]  = vv;
                if (Cbf) Cbf[idx] = f2bf(vv);
            }
        }
    }
}

// ---------------------------------------------------------------------------
// Causal attention with relative-position bias, one wave per 16-query tile.
// All matrix operands bf16 / k-major; scores & P@V & attnsum@W_rel via WMMA.
// ---------------------------------------------------------------------------
__global__ __launch_bounds__(32)
void attn_kernel(const unsigned short* __restrict__ q,
                 const unsigned short* __restrict__ k,
                 const unsigned short* __restrict__ vT,     // [B,H,DK,S]
                 const float* __restrict__ posdot,          // [B*S*H, 257]
                 const unsigned short* __restrict__ wrelT,  // [64, 288]
                 unsigned short* __restrict__ out)          // bf16 [B,S,H*DK]
{
    __shared__ float          asum[16][NRELP + 4];
    __shared__ unsigned short p_sm[16][32];
    __shared__ float          linv_sm[16];

    const int qt   = blockIdx.x;
    const int h    = blockIdx.y;
    const int b    = blockIdx.z;
    const int q0   = qt * 16;
    const int lane = threadIdx.x;
    const int half = lane >> 4;
    const int l15  = lane & 15;

    for (int i = lane; i < 16 * (NRELP + 4); i += 32) ((float*)asum)[i] = 0.f;
    __syncthreads();

    // ---- Q A-fragments (16x64 = two 16x32 chunks), b128 loads ----
    Frag16 aq[2];
    {
        const unsigned short* qp = q + ((long)(b * SS + q0 + l15)) * HD + h * DKK;
#pragma unroll
        for (int c = 0; c < 2; ++c) {
            aq[c].q[0] = *(const uint4*)(qp + c * 32 + 16 * half);
            aq[c].q[1] = *(const uint4*)(qp + c * 32 + 16 * half + 8);
        }
    }

    // ================= pass 1: row max =================
    float mx[8];
#pragma unroll
    for (int r = 0; r < 8; ++r) mx[r] = -1.0e4f;

    for (int kt = 0; kt <= q0; kt += 16) {
        v8f s = {};
#pragma unroll
        for (int c = 0; c < 2; ++c) {
            Frag16 bk;
            const unsigned short* kp =
                k + ((long)(b * SS + kt + l15)) * HD + h * DKK + c * 32 + 16 * half;
            bk.q[0] = *(const uint4*)kp;
            bk.q[1] = *(const uint4*)(kp + 8);
            s = wmma_bf16(aq[c], bk, s);
        }
        const int key = kt + l15;
#pragma unroll
        for (int r = 0; r < 8; ++r) {
            const int qq = q0 + r + 8 * half;
            const int d  = qq - key;
            float sc = -1.0e4f;
            if (d >= 0) {
                const int rel = ((d > LREL) ? LREL : d) + LREL;
                const float pd = posdot[(((long)(b * SS + qq)) * HH + h) * NREL + rel];
                sc = (s[r] + pd) * 0.125f;
            }
            mx[r] = fmaxf(mx[r], sc);
        }
    }
#pragma unroll
    for (int off = 1; off < 16; off <<= 1)
#pragma unroll
        for (int r = 0; r < 8; ++r) mx[r] = fmaxf(mx[r], __shfl_xor(mx[r], off, 16));

    // ================= pass 2: exp / accumulate =================
    float lsum[8];
#pragma unroll
    for (int r = 0; r < 8; ++r) lsum[r] = 0.f;
    v8f o[4] = {};

    for (int kt = 0; kt <= q0; kt += 32) {
#pragma unroll
        for (int t = 0; t < 2; ++t) {
            const int kb = kt + t * 16;
            v8f s = {};
#pragma unroll
            for (int c = 0; c < 2; ++c) {
                Frag16 bk;
                const unsigned short* kp =
                    k + ((long)(b * SS + kb + l15)) * HD + h * DKK + c * 32 + 16 * half;
                bk.q[0] = *(const uint4*)kp;
                bk.q[1] = *(const uint4*)(kp + 8);
                s = wmma_bf16(aq[c], bk, s);
            }
            const int key = kb + l15;
#pragma unroll
            for (int r = 0; r < 8; ++r) {
                const int m  = r + 8 * half;
                const int qq = q0 + m;
                const int d  = qq - key;
                float p = 0.f;
                if (d >= 0) {
                    const int rel = ((d > LREL) ? LREL : d) + LREL;
                    const float pd = posdot[(((long)(b * SS + qq)) * HH + h) * NREL + rel];
                    p = __expf((s[r] + pd) * 0.125f - mx[r]);
                    lsum[r] += p;
                    atomicAdd(&asum[m][rel], p);     // ds_add_f32 scatter
                }
                p_sm[m][t * 16 + l15] = f2bf(p);
            }
        }
        __syncthreads();

        // A-frag from staged P (16x32), two ds b128 reads
        Frag16 ap;
        ap.q[0] = *(const uint4*)&p_sm[l15][16 * half];
        ap.q[1] = *(const uint4*)&p_sm[l15][16 * half + 8];

        // o += P @ V : B-frags from k-major vT, b128 loads
#pragma unroll
        for (int dt = 0; dt < 4; ++dt) {
            Frag16 bv;
            const unsigned short* vp =
                vT + (((long)(b * HH + h)) * DKK + dt * 16 + l15) * (long)SS + kt + 16 * half;
            bv.q[0] = *(const uint4*)vp;
            bv.q[1] = *(const uint4*)(vp + 8);
            o[dt] = wmma_bf16(ap, bv, o[dt]);
        }
        __syncthreads();
    }

    // ---- softmax normalization ----
#pragma unroll
    for (int off = 1; off < 16; off <<= 1)
#pragma unroll
        for (int r = 0; r < 8; ++r) lsum[r] += __shfl_xor(lsum[r], off, 16);
    float linv[8];
#pragma unroll
    for (int r = 0; r < 8; ++r) linv[r] = 1.f / lsum[r];
    if (l15 == 0)
#pragma unroll
        for (int r = 0; r < 8; ++r) linv_sm[r + 8 * half] = linv[r];
    __syncthreads();

#pragma unroll
    for (int dt = 0; dt < 4; ++dt)
#pragma unroll
        for (int r = 0; r < 8; ++r) o[dt][r] *= linv[r];

    // ---- out += (attnsum/l) @ W_rel : K = 288, B-frags from k-major wrelT ----
    const float rowscale = linv_sm[l15];
    for (int rc = 0; rc < NRELP; rc += 32) {
        Frag16 ap;
#pragma unroll
        for (int i = 0; i < 8; ++i) {
            const int kk = 16 * half + 2 * i;
            ap.u[i] = pack2(asum[l15][rc + kk] * rowscale,
                            asum[l15][rc + kk + 1] * rowscale);
        }
#pragma unroll
        for (int dt = 0; dt < 4; ++dt) {
            Frag16 bw;
            const unsigned short* wp =
                wrelT + (long)(dt * 16 + l15) * NRELP + rc + 16 * half;
            bw.q[0] = *(const uint4*)wp;
            bw.q[1] = *(const uint4*)(wp + 8);
            o[dt] = wmma_bf16(ap, bw, o[dt]);
        }
    }

    // ---- store bf16 [B,S,H*DK] ----
#pragma unroll
    for (int dt = 0; dt < 4; ++dt)
#pragma unroll
        for (int r = 0; r < 8; ++r)
            out[((long)(b * SS + q0 + r + 8 * half)) * HD + h * DKK + dt * 16 + l15] =
                f2bf(o[dt][r]);
}

// ---------------------------------------------------------------------------
// LayerNorm of (a + b): y = alpha*(x-mean)/(std+eps) + bias ; optional bf16 out
// ---------------------------------------------------------------------------
__global__ __launch_bounds__(256)
void ln_kernel(const float* __restrict__ a, const float* __restrict__ bres,
               const float* __restrict__ alpha, const float* __restrict__ bias,
               float* __restrict__ outf, unsigned short* __restrict__ outbf)
{
    __shared__ float red[256], red2[256];
    const long row = blockIdx.x;
    const int  tid = threadIdx.x;
    float x[4], s = 0.f, s2 = 0.f;
#pragma unroll
    for (int i = 0; i < 4; ++i) {
        const long c = tid + i * 256;
        const float vv = a[row * DD + c] + bres[row * DD + c];
        x[i] = vv; s += vv; s2 += vv * vv;
    }
    red[tid] = s; red2[tid] = s2;
    __syncthreads();
    for (int st = 128; st > 0; st >>= 1) {
        if (tid < st) { red[tid] += red[tid + st]; red2[tid] += red2[tid + st]; }
        __syncthreads();
    }
    const float mean = red[0] * (1.f / DD);
    const float var  = fmaxf(red2[0] * (1.f / DD) - mean * mean, 0.f);
    const float inv  = 1.f / (sqrtf(var) + 1e-5f);
#pragma unroll
    for (int i = 0; i < 4; ++i) {
        const long c = tid + i * 256;
        const float y = alpha[c] * (x[i] - mean) * inv + bias[c];
        if (outf)  outf[row * DD + c]  = y;
        if (outbf) outbf[row * DD + c] = f2bf(y);
    }
}

// ---------------------------------------------------------------------------
// Host-side orchestration
// ---------------------------------------------------------------------------
extern "C" void kernel_launch(void* const* d_in, const int* in_sizes, int n_in,
                              void* d_out, int out_size, void* d_ws, size_t ws_size,
                              hipStream_t stream) {
    const float* x     = (const float*)d_in[0];
    // d_in[1] = attn_mask (causal, handled analytically)
    const float* W_q   = (const float*)d_in[2];
    const float* b_q   = (const float*)d_in[3];
    const float* W_k   = (const float*)d_in[4];
    const float* b_k   = (const float*)d_in[5];
    const float* W_v   = (const float*)d_in[6];
    const float* b_v   = (const float*)d_in[7];
    const float* W_o   = (const float*)d_in[8];
    const float* b_o   = (const float*)d_in[9];
    const float* W_rel = (const float*)d_in[10];
    const float* ln1a  = (const float*)d_in[11];
    const float* ln1b  = (const float*)d_in[12];
    const float* ln2a  = (const float*)d_in[13];
    const float* ln2b  = (const float*)d_in[14];
    const float* W1    = (const float*)d_in[15];
    const float* b1    = (const float*)d_in[16];
    const float* W2    = (const float*)d_in[17];
    const float* b2    = (const float*)d_in[18];
    float* outp = (float*)d_out;

    const long M = (long)BB * SS;                          // 4096 token rows

    // ---- workspace layout: f32 region first, then bf16 (ushort) region ----
    float* fp = (float*)d_ws;
    size_t fo = 0;
    float* pdbuf = fp + fo; fo += (size_t)M * HH * NREL;   // posdot
    float* vf32  = fp + fo; fo += (size_t)M * DD;          // V pre-transpose
    float* proj  = fp + fo; fo += (size_t)M * DD;
    float* hf32  = fp + fo; fo += (size_t)M * DD;
    float* f2    = fp + fo; fo += (size_t)M * DD;

    unsigned short* up = (unsigned short*)(fp + fo);
    size_t uo = 0;
    unsigned short* xbf    = up + uo; uo += (size_t)M * DD;
    unsigned short* wqbf   = up + uo; uo += (size_t)DD * DD;
    unsigned short* wkbf   = up + uo; uo += (size_t)DD * DD;
    unsigned short* wvbf   = up + uo; uo += (size_t)DD * DD;
    unsigned short* wobf   = up + uo; uo += (size_t)DD * DD;
    unsigned short* w1bf   = up + uo; uo += (size_t)DFFN * DD;
    unsigned short* w2bf   = up + uo; uo += (size_t)DD * DFFN;
    unsigned short* wrelbf = up + uo; uo += (size_t)NREL * DKK;    // 16448 (x8 aligned)
    unsigned short* wrelT  = up + uo; uo += (size_t)DKK * NRELP;   // 18432
    unsigned short* qbf    = up + uo; uo += (size_t)M * DD;
    unsigned short* kbf    = up + uo; uo += (size_t)M * DD;
    unsigned short* vTb    = up + uo; uo += (size_t)M * DD;
    unsigned short* attbf  = up + uo; uo += (size_t)M * DD;
    unsigned short* hbf    = up + uo; uo += (size_t)M * DD;
    unsigned short* ffnbf  = up + uo; uo += (size_t)M * DFFN;

    const dim3 blk(256);
    auto cvt = [&](const float* src, unsigned short* dst, long n) {
        cvt_bf16_kernel<<<dim3((unsigned)((n + 255) / 256)), blk, 0, stream>>>(src, dst, n);
    };

    // ---- one-time bf16 conversions (HBM-bandwidth trivial vs GEMM FLOPs) ----
    cvt(x,    xbf,  M * DD);
    cvt(W_q,  wqbf, (long)DD * DD);
    cvt(W_k,  wkbf, (long)DD * DD);
    cvt(W_v,  wvbf, (long)DD * DD);
    cvt(W_o,  wobf, (long)DD * DD);
    cvt(W1,   w1bf, (long)DFFN * DD);
    cvt(W2,   w2bf, (long)DD * DFFN);
    cvt(W_rel, wrelbf, (long)NREL * DKK);
    wrelT_kernel<<<dim3((DKK * NRELP + 255) / 256), blk, 0, stream>>>(W_rel, wrelT);

    // GEMM grids: block tile 64(M) x 128(N)
    const dim3 gProj((DD + 127) / 128, M / 64);
    const dim3 gPd((NREL + 127) / 128, (M * HH) / 64);
    const dim3 gFfn((DFFN + 127) / 128, M / 64);

    // QKV projections
    gemm_bf16_kernel<<<gProj, blk, 0, stream>>>(xbf, wqbf, b_q, nullptr, qbf,  M, DD, DD, 0);
    gemm_bf16_kernel<<<gProj, blk, 0, stream>>>(xbf, wkbf, b_k, nullptr, kbf,  M, DD, DD, 0);
    gemm_bf16_kernel<<<gProj, blk, 0, stream>>>(xbf, wvbf, b_v, vf32,   nullptr, M, DD, DD, 0);
    vT_kernel<<<dim3((unsigned)(M * HH * DKK / 64)), blk, 0, stream>>>(vf32, vTb);

    // posdot[(b,s,h), rel] = q . W_rel[rel]
    gemm_bf16_kernel<<<gPd, blk, 0, stream>>>(qbf, wrelbf, nullptr, pdbuf, nullptr,
                                              (int)(M * HH), NREL, DKK, 0);

    // causal attention + rel-pos (score & value side)
    attn_kernel<<<dim3(SS / 16, HH, BB), dim3(32), 0, stream>>>(
        qbf, kbf, vTb, pdbuf, wrelT, attbf);

    // output projection, LN1
    gemm_bf16_kernel<<<gProj, blk, 0, stream>>>(attbf, wobf, b_o, proj, nullptr, M, DD, DD, 0);
    ln_kernel<<<dim3((unsigned)M), blk, 0, stream>>>(x, proj, ln1a, ln1b, hf32, hbf);

    // FFN + LN2
    gemm_bf16_kernel<<<gFfn, blk, 0, stream>>>(hbf, w1bf, b1, nullptr, ffnbf, M, DFFN, DD, 1);
    gemm_bf16_kernel<<<gProj, blk, 0, stream>>>(ffnbf, w2bf, b2, f2, nullptr, M, DD, DFFN, 0);
    ln_kernel<<<dim3((unsigned)M), blk, 0, stream>>>(hf32, f2, ln2a, ln2b, outp, nullptr);
}
